// DetectorHelper_34428457844904
// MI455X (gfx1250) — matrix operand order repaired and checked by reading, not compiled, not run on hardware
//
#include <hip/hip_runtime.h>

// ---------------- problem constants ----------------
#define BB   128
#define TT   1024
#define FF   64
#define HH   256
#define NWG  16          // j-tiles of 16 h-columns each
#define WHH_LD 264       // padded LDS stride for K=256 (bank-conflict avoidance)
#define WIH_LD 72        // padded LDS stride for K=64

typedef __bf16 bf16_t;
typedef __attribute__((ext_vector_type(16))) __bf16 v16bf;
typedef __attribute__((ext_vector_type(8)))  __bf16 v8bf;
typedef __attribute__((ext_vector_type(8)))  float  v8f;

__device__ __forceinline__ float sigm_f(float x) {
  return 1.0f / (1.0f + __expf(-x));
}
__device__ __forceinline__ float tanh_f(float x) {
  float xc = fminf(fmaxf(x, -15.0f), 15.0f);
  float e = __expf(2.0f * xc);
  return (e - 1.0f) / (e + 1.0f);
}

// A-fragment (16x32 bf16, MxK) from row-major bf16 matrix.
// ISA layout: lanes 0-15 -> M=lane, elems 0-7 = K k0..k0+7, elems 8-15 = K k0+16..k0+23
//             lanes 16-31 -> same M, K offset +8 (and +24).
__device__ __forceinline__ v16bf load_a_bf16(const bf16_t* base, int ld, int row0,
                                             int k0, int lane) {
  int r = lane & 15, half = lane >> 4;
  const bf16_t* p = base + (size_t)(row0 + r) * ld + k0 + half * 8;
  v8bf lo = *(const v8bf*)p;
  v8bf hi = *(const v8bf*)(p + 16);
  return __builtin_shufflevector(lo, hi, 0,1,2,3,4,5,6,7,8,9,10,11,12,13,14,15);
}

// Same A layout but sourced from fp32 (the raw time-series input), converted to bf16.
__device__ __forceinline__ v16bf load_a_f32(const float* base, int ld, int row0,
                                            int k0, int lane) {
  int r = lane & 15, half = lane >> 4;
  const float* p = base + (size_t)(row0 + r) * ld + k0 + half * 8;
  v8f lo = *(const v8f*)p;
  v8f hi = *(const v8f*)(p + 16);
  v16bf a;
#pragma unroll
  for (int i = 0; i < 8; ++i) { a[i] = (__bf16)lo[i]; a[i + 8] = (__bf16)hi[i]; }
  return a;
}

// B-fragment (32x16 bf16, KxN) from LDS slice w[c][k] (row = output column, k contiguous).
// ISA layout: lanes 0-15 -> N=lane, elems 0-7 = K k0..k0+7, elems 8-15 = K k0+16..k0+23; lanes 16-31 -> K offset +8 (and +24).
__device__ __forceinline__ v16bf load_b_lds(const bf16_t* w, int ld, int k0, int lane) {
  int c = lane & 15, half = lane >> 4;
  const bf16_t* p = w + c * ld + k0 + half * 8;
  v8bf lo = *(const v8bf*)p;
  v8bf hi = *(const v8bf*)(p + 16);
  return __builtin_shufflevector(lo, hi, 0,1,2,3,4,5,6,7,8,9,10,11,12,13,14,15);
}

__global__ void lstm_ae_init(int* ctr) {
  if (threadIdx.x == 0) *ctr = 0;
}

__global__ void __launch_bounds__(256, 1)
lstm_ae_kernel(const float* __restrict__ X,
               const float* __restrict__ Wih_e, const float* __restrict__ Whh_e,
               const float* __restrict__ b_e,
               const float* __restrict__ Wih_d, const float* __restrict__ Whh_d,
               const float* __restrict__ b_d,
               const float* __restrict__ Wout,  const float* __restrict__ bout,
               float* __restrict__ out, int* __restrict__ ctr, bf16_t* __restrict__ hbuf) {
  // Persistent LDS weight slices for this workgroup's 16 h-columns (all 4 gates).
  __shared__ bf16_t s_whh[4][16][WHH_LD];   // 33 KB: recurrent weights (current phase)
  __shared__ bf16_t s_wih[4][16][WIH_LD];   //  9 KB: input weights (current phase)
  __shared__ bf16_t s_wout[16][WHH_LD];     //  8 KB: output projection (decoder, WGs 0-3)

  const int wg   = blockIdx.x;     // j-tile: h-columns [wg*16, wg*16+16)
  const int tid  = threadIdx.x;
  const int wave = tid >> 5;       // M-tile: batch rows [wave*16, wave*16+16)
  const int lane = tid & 31;
  const int n    = lane & 15;      // column-within-tile for B/C/D fragments
  const int half = lane >> 4;
  const int m0   = wave * 16;

  // ---- load-phase helpers ----
  auto load_whh = [&](const float* W) {
    for (int idx = tid; idx < 4 * 16 * HH; idx += 256) {
      int g = idx >> 12, rem = idx & 4095, c = rem >> 8, k = rem & 255;
      s_whh[g][c][k] = (bf16_t)W[(size_t)(g * HH + wg * 16 + c) * HH + k];
    }
  };
  auto load_wih = [&](const float* W) {
    for (int idx = tid; idx < 4 * 16 * FF; idx += 256) {
      int g = idx >> 10, rem = idx & 1023, c = rem >> 6, k = rem & 63;
      s_wih[g][c][k] = (bf16_t)W[(size_t)(g * HH + wg * 16 + c) * FF + k];
    }
  };

  // Per-lane gate biases for this wave's columns (same for all 8 M rows in a VGPR slot).
  float be[4], bd[4];
#pragma unroll
  for (int g = 0; g < 4; ++g) {
    be[g] = b_e[g * HH + wg * 16 + n];
    bd[g] = b_d[g * HH + wg * 16 + n];
  }
  float bo = (wg < 4) ? bout[wg * 16 + n] : 0.0f;

  // ---- grid barrier (monotonic counter, device scope) ----
  int ep = 0;
  auto gbar = [&]() {
    __syncthreads();
    __threadfence();
    if (tid == 0) {
      __hip_atomic_fetch_add(ctr, 1, __ATOMIC_RELEASE, __HIP_MEMORY_SCOPE_AGENT);
      const int target = (ep + 1) * NWG;
      while (__hip_atomic_load(ctr, __ATOMIC_ACQUIRE, __HIP_MEMORY_SCOPE_AGENT) < target)
        __builtin_amdgcn_s_sleep(2);
    }
    __syncthreads();
    ++ep;
  };

  // ---- init: encoder weights to LDS, zero h buffer 0 (our columns) ----
  load_whh(Whh_e);
  load_wih(Wih_e);
  for (int idx = tid; idx < BB * 16; idx += 256) {
    int m = idx >> 4, c = idx & 15;
    hbuf[(size_t)m * HH + wg * 16 + c] = (bf16_t)0.0f;
  }
  gbar();  // episode 0: h0 visible everywhere

  // Cell state c lives in registers for the whole 2048-step recurrence.
  v8f c_state;
#pragma unroll
  for (int v = 0; v < 8; ++v) c_state[v] = 0.0f;

  int cur = 0;

  // One recurrent step: gates = x_t@Wih^T + h@Whh^T + b; LSTM update; h -> hbuf[cur^1].
  // If emit: also out_t = h@Wout^T + b_out (uses PRE-update h), written by WGs 0-3.
  auto step = [&](const float* bias4, int t, bool emit) {
    const bf16_t* hcur = hbuf + (size_t)cur * (BB * HH);
    bf16_t*       hnxt = hbuf + (size_t)(cur ^ 1) * (BB * HH);

    v8f acc[4];
#pragma unroll
    for (int g = 0; g < 4; ++g) {
      float bg = bias4[g];
#pragma unroll
      for (int v = 0; v < 8; ++v) acc[g][v] = bg;
    }

    // x contribution: K = 64 (fp32 -> bf16 on the fly)
#pragma unroll
    for (int k0 = 0; k0 < FF; k0 += 32) {
      v16bf a = load_a_f32(X + (size_t)t * FF, TT * FF, m0, k0, lane);
#pragma unroll
      for (int g = 0; g < 4; ++g) {
        v16bf b = load_b_lds(&s_wih[g][0][0], WIH_LD, k0, lane);
        acc[g] = __builtin_amdgcn_wmma_f32_16x16x32_bf16(
            false, a, false, b, (short)0, acc[g], false, false);
      }
    }

    v8f aout;
    if (emit && wg < 4) {
#pragma unroll
      for (int v = 0; v < 8; ++v) aout[v] = bo;
    }

    // h contribution: K = 256; decoder fuses the output projection on the same A-frags
#pragma unroll
    for (int k0 = 0; k0 < HH; k0 += 32) {
      v16bf a = load_a_bf16(hcur, HH, m0, k0, lane);
#pragma unroll
      for (int g = 0; g < 4; ++g) {
        v16bf b = load_b_lds(&s_whh[g][0][0], WHH_LD, k0, lane);
        acc[g] = __builtin_amdgcn_wmma_f32_16x16x32_bf16(
            false, a, false, b, (short)0, acc[g], false, false);
      }
      if (emit && wg < 4) {
        v16bf b = load_b_lds(&s_wout[0][0], WHH_LD, k0, lane);
        aout = __builtin_amdgcn_wmma_f32_16x16x32_bf16(
            false, a, false, b, (short)0, aout, false, false);
      }
    }

    if (emit && wg < 4) {
#pragma unroll
      for (int v = 0; v < 8; ++v) {
        int m = m0 + v + half * 8;
        out[(size_t)m * (TT * FF) + (size_t)t * FF + wg * 16 + n] = aout[v];
      }
    }

    // Elementwise LSTM update — all four gates for (m, j) sit in the same lane/slot.
#pragma unroll
    for (int v = 0; v < 8; ++v) {
      float ig = sigm_f(acc[0][v]);
      float fg = sigm_f(acc[1][v]);
      float gg = tanh_f(acc[2][v]);
      float og = sigm_f(acc[3][v]);
      float cc = fg * c_state[v] + ig * gg;
      c_state[v] = cc;
      float hh = og * tanh_f(cc);
      int m = m0 + v + half * 8;
      hnxt[(size_t)m * HH + wg * 16 + n] = (bf16_t)hh;
    }
  };

  // ---- encoder: t = 0..T-1 ----
  for (int t = 0; t < TT; ++t) {
    if (t + 1 < TT)  // warm next timestep's input slice into cache
      __builtin_prefetch(X + (size_t)(m0 + n) * (TT * FF) + (size_t)(t + 1) * FF, 0, 0);
    step(be, t, false);
    gbar();
    cur ^= 1;
  }

  // ---- phase switch: decoder weights into LDS (safe: gbar synced the block) ----
  load_whh(Whh_d);
  load_wih(Wih_d);
  if (wg < 4) {
    for (int idx = tid; idx < 16 * HH; idx += 256) {
      int c = idx >> 8, k = idx & 255;
      s_wout[c][k] = (bf16_t)Wout[(size_t)(wg * 16 + c) * HH + k];
    }
  }
  __syncthreads();

  // ---- decoder: s = 0..T-1, x index and output index both T-1-s ----
  for (int s = 0; s < TT; ++s) {
    int t = TT - 1 - s;
    step(bd, t, true);
    gbar();
    cur ^= 1;
  }
}

extern "C" void kernel_launch(void* const* d_in, const int* in_sizes, int n_in,
                              void* d_out, int out_size, void* d_ws, size_t ws_size,
                              hipStream_t stream) {
  const float* X     = (const float*)d_in[0];
  const float* Wih_e = (const float*)d_in[1];
  const float* Whh_e = (const float*)d_in[2];
  const float* b_e   = (const float*)d_in[3];
  const float* Wih_d = (const float*)d_in[4];
  const float* Whh_d = (const float*)d_in[5];
  const float* Whh_b = (const float*)d_in[6];
  const float* Wout  = (const float*)d_in[7];
  const float* bout  = (const float*)d_in[8];

  // Workspace layout: [0,256) barrier counter; then bf16 h double-buffer (128 KB).
  const size_t need = 256 + (size_t)2 * BB * HH * sizeof(bf16_t);
  if (ws_size < need) return;
  int*    ctr  = (int*)d_ws;
  bf16_t* hbuf = (bf16_t*)((char*)d_ws + 256);

  lstm_ae_init<<<dim3(1), dim3(1), 0, stream>>>(ctr);
  lstm_ae_kernel<<<dim3(NWG), dim3(256), 0, stream>>>(
      X, Wih_e, Whh_e, b_e, Wih_d, Whh_d, Whh_b, Wout, bout,
      (float*)d_out, ctr, hbuf);
}
